// BigBirdSiameseNetwork_12841952215099
// MI455X (gfx1250) — matrix operand order repaired and checked by reading, not compiled, not run on hardware
//
#include <hip/hip_runtime.h>
#include <hip/hip_fp16.h>

// ---------------- model constants ----------------
constexpr int kD   = 768;     // hidden
constexpr int kNH  = 12;      // heads
constexpr int kHD  = 64;      // head dim
constexpr int kBS  = 64;      // BigBird block size
constexpr int kNL  = 2;       // layers
constexpr int kFF  = 3072;    // FFN inner
constexpr int kS   = 4096;    // seq len
constexpr int kB   = 2;       // batch
constexpr int kNB  = kS / kBS;   // 64 blocks
constexpr int kR   = 3;          // random blocks
constexpr int kTok = kB * kS;    // 8192 tokens

typedef __attribute__((ext_vector_type(16))) _Float16 v16h;
typedef __attribute__((ext_vector_type(8)))  float    v8f;

union Frag { v16h h; unsigned u[8]; };

// A-matrix (16x32 f16) half-pair K index for VGPR i, lane-half hi
__device__ __forceinline__ int kmapA(int i, int hi) {
  return ((i < 4) ? 2 * i : 2 * i + 8) + hi * 8;
}
// B-matrix (32x16 f16) half-pair K index for VGPR i, lane-half hi
__device__ __forceinline__ int kmapB(int i, int hi) {
  return ((i < 4) ? 2 * i : 2 * i + 8) + hi * 8;
}

// ---- CDNA5 async memory->LDS copy (ASYNCcnt tracked, no VGPR transit) ----
// LDS operand: low 32 bits of the generic shared pointer are the wave-relative
// LDS byte offset (ISA 10.2).
__device__ __forceinline__ void async_load_b128(unsigned lds_off,
                                                unsigned long long gaddr) {
  asm volatile("global_load_async_to_lds_b128 %0, %1, off"
               :: "v"(lds_off), "v"(gaddr) : "memory");
}
__device__ __forceinline__ void async_wait() {
  asm volatile("s_wait_asynccnt 0x0" ::: "memory");
}

// ---------------- small utility kernels ----------------
__global__ void f32_to_f16_kernel(const float* __restrict__ src,
                                  __half* __restrict__ dst, int n) {
  int i = blockIdx.x * blockDim.x + threadIdx.x;
  int stride = gridDim.x * blockDim.x;
  for (; i < n; i += stride) dst[i] = __float2half(src[i]);
}

// key-block table: edge query blocks attend to all blocks; middle blocks to
// [0, nb-1, qb-1, qb, qb+1, rand0..2] (duplicates kept, matching reference)
__global__ void build_kb_kernel(const int* __restrict__ rand_blocks,
                                int* __restrict__ kb_idx, int* __restrict__ kb_cnt) {
  int qb = threadIdx.x;
  if (qb >= kNB) return;
  if (qb == 0 || qb == kNB - 1) {
    kb_cnt[qb] = kNB;
    for (int i = 0; i < kNB; ++i) kb_idx[qb * kNB + i] = i;
  } else {
    kb_cnt[qb] = 5 + kR;
    int* p = kb_idx + qb * kNB;
    p[0] = 0; p[1] = kNB - 1; p[2] = qb - 1; p[3] = qb; p[4] = qb + 1;
    for (int r = 0; r < kR; ++r) p[5 + r] = rand_blocks[(qb - 1) * kR + r];
  }
}

__device__ __forceinline__ float block_reduce_sum(float x, float* red) {
  int t = threadIdx.x;
  red[t] = x; __syncthreads();
  for (int o = 128; o > 0; o >>= 1) {
    if (t < o) red[t] += red[t + o];
    __syncthreads();
  }
  float r = red[0];
  __syncthreads();
  return r;
}

// embeddings + LayerNorm; one block per token, 256 threads * 3 features
__global__ void embed_ln_kernel(const int* __restrict__ ids, const int* __restrict__ tt,
                                const float* __restrict__ wemb, const float* __restrict__ temb,
                                const float* __restrict__ pemb,
                                const float* __restrict__ g, const float* __restrict__ bta,
                                float* __restrict__ h, __half* __restrict__ h16) {
  __shared__ float red[256];
  int tok = blockIdx.x;
  int s = tok % kS;
  int t = threadIdx.x;
  int id = ids[tok], ty = tt[tok];
  float v[3]; float sum = 0.f;
  for (int i = 0; i < 3; ++i) {
    int d = t + i * 256;
    v[i] = wemb[(size_t)id * kD + d] + temb[(size_t)ty * kD + d] + pemb[(size_t)s * kD + d];
    sum += v[i];
  }
  float mu = block_reduce_sum(sum, red) / kD;
  float sq = 0.f;
  for (int i = 0; i < 3; ++i) { float d0 = v[i] - mu; sq += d0 * d0; }
  float var = block_reduce_sum(sq, red) / kD;
  float rs = rsqrtf(var + 1e-12f);
  for (int i = 0; i < 3; ++i) {
    int d = t + i * 256;
    float o = g[d] * (v[i] - mu) * rs + bta[d];
    h[(size_t)tok * kD + d] = o;
    h16[(size_t)tok * kD + d] = __float2half(o);
  }
}

// h = LN(h + add); also refresh f16 shadow
__global__ void add_ln_kernel(float* __restrict__ h, const float* __restrict__ add,
                              const float* __restrict__ g, const float* __restrict__ bta,
                              __half* __restrict__ h16) {
  __shared__ float red[256];
  int tok = blockIdx.x;
  int t = threadIdx.x;
  float v[3]; float sum = 0.f;
  for (int i = 0; i < 3; ++i) {
    int d = t + i * 256;
    float x = h[(size_t)tok * kD + d] + add[(size_t)tok * kD + d];
    v[i] = x; sum += x;
  }
  float mu = block_reduce_sum(sum, red) / kD;
  float sq = 0.f;
  for (int i = 0; i < 3; ++i) { float d0 = v[i] - mu; sq += d0 * d0; }
  float var = block_reduce_sum(sq, red) / kD;
  float rs = rsqrtf(var + 1e-12f);
  for (int i = 0; i < 3; ++i) {
    int d = t + i * 256;
    float o = g[d] * (v[i] - mu) * rs + bta[d];
    h[(size_t)tok * kD + d] = o;
    h16[(size_t)tok * kD + d] = __float2half(o);
  }
}

// ---------------- WMMA GEMM ----------------
// C[M,N] = act(A[M,K](f16, lda) * Bw[K,N](f16) + bias[N])
// block tile 128x128, 256 threads = 8 waves, each wave 32x64 (2x4 WMMA tiles)
// A tile staged via async global->LDS DMA; B tile staged transposed manually.
// ACT: 0=none, 1=exact GELU, 2=tanh.  OUT_HALF: store f16 else f32.
template <int ACT, bool OUT_HALF>
__global__ __launch_bounds__(256, 1)
void gemm_kernel(const __half* __restrict__ A, const __half* __restrict__ Bw,
                 const float* __restrict__ bias, void* __restrict__ Cout,
                 int M, int N, int K, int lda) {
  __shared__ __attribute__((aligned(16))) __half As[128 * 32];  // [row][k]
  __shared__ __attribute__((aligned(16))) __half Bs[128 * 32];  // [n][k] (transposed)
  const unsigned* AsU = (const unsigned*)As;
  const unsigned* BsU = (const unsigned*)Bs;
  unsigned ldsA = (unsigned)(uintptr_t)(void*)As;
  const char* Abytes = (const char*)A;

  int t = threadIdx.x;
  int wave = t >> 5, lane = t & 31, l15 = lane & 15, hi = (lane >> 4) & 1;
  int waveM = wave >> 1, waveN = wave & 1;
  int m0 = blockIdx.y * 128;
  int n0 = blockIdx.x * 128;

  v8f acc[2][4] = {};

  for (int k0 = 0; k0 < K; k0 += 32) {
    // stage A tile (128 rows x 64B) via async DMA: 512 x b128, 2 per thread.
    // Rows >= M are skipped; stale LDS only feeds never-stored output rows.
#pragma unroll
    for (int i = 0; i < 2; ++i) {
      int f = t * 2 + i;        // b128 chunk index
      int row = f >> 2;         // 4 chunks per row
      int cib = f & 3;
      if (m0 + row < M)
        async_load_b128(ldsA + f * 16,
                        (unsigned long long)(uintptr_t)(Abytes +
                            ((size_t)(m0 + row) * (size_t)lda + (size_t)k0) * 2 + cib * 16));
    }
    // stage B tile transposed: Bs[n][k] = Bw[k0+k][n0+n]  (32x128 halves)
#pragma unroll
    for (int i = 0; i < 16; ++i) {
      int f = t * 16 + i;       // half index over 32*128
      int k = f >> 7, n = f & 127;
      Bs[n * 32 + k] = Bw[(size_t)(k0 + k) * N + (n0 + n)];
    }
    // prefetch next B K-chunk while this one is consumed
    if (k0 + 32 < K && t < 64)
      __builtin_prefetch(&Bw[(size_t)(k0 + 32 + (t & 31)) * N + n0 + ((t >> 5) ? 64 : 0)], 0, 1);
    async_wait();
    __syncthreads();

    Frag aF[2];
#pragma unroll
    for (int mt = 0; mt < 2; ++mt) {
      int row = waveM * 32 + mt * 16 + l15;
#pragma unroll
      for (int i = 0; i < 8; ++i) {
        int kk = kmapA(i, hi);
        aF[mt].u[i] = AsU[row * 16 + kk / 2];
      }
    }
#pragma unroll
    for (int nt = 0; nt < 4; ++nt) {
      Frag bF;
      int col = waveN * 64 + nt * 16 + l15;
#pragma unroll
      for (int i = 0; i < 8; ++i) {
        int kk = kmapB(i, hi);
        bF.u[i] = BsU[col * 16 + kk / 2];
      }
#pragma unroll
      for (int mt = 0; mt < 2; ++mt)
        acc[mt][nt] = __builtin_amdgcn_wmma_f32_16x16x32_f16(
            false, aF[mt].h, false, bF.h, (short)0, acc[mt][nt], false, false);
    }
    __syncthreads();
  }

  // epilogue (C layout: lane holds col=l15, rows r + 8*hi)
#pragma unroll
  for (int mt = 0; mt < 2; ++mt) {
#pragma unroll
    for (int nt = 0; nt < 4; ++nt) {
#pragma unroll
      for (int r = 0; r < 8; ++r) {
        int row = m0 + waveM * 32 + mt * 16 + r + 8 * hi;
        int col = n0 + waveN * 64 + nt * 16 + l15;
        if (row < M) {
          float x = acc[mt][nt][r] + bias[col];
          if (ACT == 1) x = 0.5f * x * (1.0f + erff(x * 0.70710678118f));
          else if (ACT == 2) x = tanhf(x);
          if (OUT_HALF) ((__half*)Cout)[(size_t)row * N + col] = __float2half(x);
          else          ((float*)Cout)[(size_t)row * N + col] = x;
        }
      }
    }
  }
}

// ---------------- block-sparse flash attention ----------------
// grid: B*NH*NB blocks, 128 threads (4 waves); each wave owns 16 query rows.
// K block staged via async global->LDS DMA; V staged with manual transpose.
__global__ __launch_bounds__(128, 1)
void attn_kernel(const __half* __restrict__ q16, const __half* __restrict__ k16,
                 const __half* __restrict__ v16, const int* __restrict__ amask,
                 const int* __restrict__ kb_idx, const int* __restrict__ kb_cnt,
                 __half* __restrict__ ctx16) {
  __shared__ __attribute__((aligned(16))) __half Ks[kBS * kHD];    // [key][d]
  __shared__ __attribute__((aligned(16))) __half Vt[kHD * kBS];    // [d][key]
  __shared__ __attribute__((aligned(16))) __half Ps[4][16 * kHD];  // per-wave P

  int bid = blockIdx.x;
  int qb = bid % kNB;
  int head = (bid / kNB) % kNH;
  int b = bid / (kNB * kNH);
  int t = threadIdx.x;
  int wave = t >> 5, lane = t & 31, l15 = lane & 15, hi = (lane >> 4) & 1;

  size_t base = (size_t)b * kS * kD + (size_t)head * kHD;  // + tok*kD + d

  // Q fragments (A layout), 2 K-chunks of 32 over HD=64
  Frag aQ[2];
  {
    const unsigned* Qu = (const unsigned*)q16;
    int qtok = qb * kBS + wave * 16 + l15;
#pragma unroll
    for (int c = 0; c < 2; ++c)
#pragma unroll
      for (int i = 0; i < 8; ++i) {
        int d = kmapA(i, hi) + c * 32;
        aQ[c].u[i] = Qu[(base + (size_t)qtok * kD + d) / 2];
      }
  }

  float mrow[8], lrow[8];
#pragma unroll
  for (int r = 0; r < 8; ++r) { mrow[r] = -1e30f; lrow[r] = 0.f; }
  v8f O[4] = {};

  int cnt = kb_cnt[qb];
  const char* Kbytes = (const char*)k16;
  const __half2* V2 = (const __half2*)v16;
  unsigned ldsK = (unsigned)(uintptr_t)(void*)Ks;
  const unsigned* KsC = (const unsigned*)Ks;
  const unsigned* VtU = (const unsigned*)Vt;

  for (int g = 0; g < cnt; ++g) {
    int kb = kb_idx[qb * kNB + g];
    // stage K block (64 rows x 128B) via async DMA: 512 x b128, 4 per thread
#pragma unroll
    for (int i = 0; i < 4; ++i) {
      int f = t * 4 + i;         // b128 chunk index
      int key = f >> 3;          // 8 chunks per key row
      int cib = f & 7;
      async_load_b128(ldsK + f * 16,
                      (unsigned long long)(uintptr_t)(Kbytes +
                          (base + (size_t)(kb * kBS + key) * kD) * 2 + cib * 16));
    }
    // stage V transposed [d][key] via half2 loads (2048 pairs / 128 threads)
#pragma unroll
    for (int i = 0; i < 16; ++i) {
      int f = i * 128 + t;
      int key = f >> 5;                 // 32 pairs per key row
      int d2 = f & 31;
      __half2 pv = V2[(base + (size_t)(kb * kBS + key) * kD) / 2 + d2];
      Vt[(2 * d2) * kBS + key]     = __low2half(pv);
      Vt[(2 * d2 + 1) * kBS + key] = __high2half(pv);
    }
    // prefetch next key block (each thread pair covers one key row of K and V)
    if (g + 1 < cnt) {
      int nkb = kb_idx[qb * kNB + g + 1];
      size_t prow = base + (size_t)(nkb * kBS + (t >> 1)) * kD + (t & 1) * 32;
      __builtin_prefetch(&k16[prow], 0, 1);
      __builtin_prefetch(&v16[prow], 0, 1);
    }
    async_wait();
    __syncthreads();

    // S = Q K^T   (B operand = Ks rows: B[kdim=d][n=key])
    v8f sc[4];
#pragma unroll
    for (int j = 0; j < 4; ++j) {
      v8f s = {};
#pragma unroll
      for (int c = 0; c < 2; ++c) {
        Frag bK;
#pragma unroll
        for (int i = 0; i < 8; ++i) {
          int d = kmapB(i, hi) + c * 32;
          bK.u[i] = KsC[(j * 16 + l15) * 32 + d / 2];
        }
        s = __builtin_amdgcn_wmma_f32_16x16x32_f16(
            false, aQ[c].h, false, bK.h, (short)0, s, false, false);
      }
      sc[j] = s;
    }

    // scale + additive mask
    float madd[4];
#pragma unroll
    for (int j = 0; j < 4; ++j) {
      int key = kb * kBS + j * 16 + l15;
      madd[j] = (1.0f - (float)amask[b * kS + key]) * -1e9f;
    }
#pragma unroll
    for (int j = 0; j < 4; ++j)
#pragma unroll
      for (int r = 0; r < 8; ++r) sc[j][r] = sc[j][r] * 0.125f + madd[j];

    // online softmax update (rows live across 16 lanes sharing lane-half)
#pragma unroll
    for (int r = 0; r < 8; ++r) {
      float mx = fmaxf(fmaxf(sc[0][r], sc[1][r]), fmaxf(sc[2][r], sc[3][r]));
#pragma unroll
      for (int o = 1; o < 16; o <<= 1) mx = fmaxf(mx, __shfl_xor(mx, o, 32));
      float nm = fmaxf(mrow[r], mx);
      float corr = __expf(mrow[r] - nm);
      float rs0 = 0.f;
#pragma unroll
      for (int j = 0; j < 4; ++j) {
        float p = __expf(sc[j][r] - nm);
        sc[j][r] = p; rs0 += p;
      }
#pragma unroll
      for (int o = 1; o < 16; o <<= 1) rs0 += __shfl_xor(rs0, o, 32);
      lrow[r] = lrow[r] * corr + rs0;
      mrow[r] = nm;
#pragma unroll
      for (int j = 0; j < 4; ++j) O[j][r] *= corr;
    }

    // P (C layout) -> LDS -> A-layout fragments (same-wave LDS ops are in-order)
    __half* Pw = Ps[wave];
#pragma unroll
    for (int j = 0; j < 4; ++j)
#pragma unroll
      for (int r = 0; r < 8; ++r)
        Pw[(r + 8 * hi) * kHD + j * 16 + l15] = __float2half(sc[j][r]);

    const unsigned* Pu = (const unsigned*)Ps[wave];
    Frag pF[2];
#pragma unroll
    for (int c = 0; c < 2; ++c)
#pragma unroll
      for (int i = 0; i < 8; ++i) {
        int kk = kmapA(i, hi) + c * 32;
        pF[c].u[i] = Pu[l15 * 32 + kk / 2];
      }

    // O += P V   (B operand = Vt: B[kdim=key][n=d], Vt stored [d][key])
#pragma unroll
    for (int j = 0; j < 4; ++j) {
#pragma unroll
      for (int c = 0; c < 2; ++c) {
        Frag bV;
#pragma unroll
        for (int i = 0; i < 8; ++i) {
          int kk = kmapB(i, hi) + c * 32;
          bV.u[i] = VtU[(j * 16 + l15) * 32 + kk / 2];
        }
        O[j] = __builtin_amdgcn_wmma_f32_16x16x32_f16(
            false, pF[c].h, false, bV.h, (short)0, O[j], false, false);
      }
    }
    __syncthreads();
  }

  // finalize: divide by row sum, write ctx in [B,S,D] (feature = head*64 + d)
#pragma unroll
  for (int r = 0; r < 8; ++r) {
    float inv = 1.0f / lrow[r];
    int tok = qb * kBS + wave * 16 + r + 8 * hi;
#pragma unroll
    for (int j = 0; j < 4; ++j)
      ctx16[base + (size_t)tok * kD + j * 16 + l15] = __float2half(O[j][r] * inv);
  }
}

// ---------------- host orchestration ----------------
extern "C" void kernel_launch(void* const* d_in, const int* in_sizes, int n_in,
                              void* d_out, int out_size, void* d_ws, size_t ws_size,
                              hipStream_t stream) {
  (void)in_sizes; (void)n_in; (void)out_size; (void)ws_size;
  const int*   input_ids  = (const int*)d_in[0];
  const int*   attn_mask  = (const int*)d_in[1];
  const int*   token_type = (const int*)d_in[2];
  const int*   rand_blks  = (const int*)d_in[3];
  const float* word_emb   = (const float*)d_in[4];
  const float* type_emb   = (const float*)d_in[5];
  const float* pos_emb    = (const float*)d_in[6];
  const float* emb_ln_s   = (const float*)d_in[7];
  const float* emb_ln_b   = (const float*)d_in[8];
  const float* Wq = (const float*)d_in[9];   const float* bq = (const float*)d_in[10];
  const float* Wk = (const float*)d_in[11];  const float* bk = (const float*)d_in[12];
  const float* Wv = (const float*)d_in[13];  const float* bv = (const float*)d_in[14];
  const float* Wo = (const float*)d_in[15];  const float* bo = (const float*)d_in[16];
  const float* ln1_s = (const float*)d_in[17]; const float* ln1_b = (const float*)d_in[18];
  const float* W1 = (const float*)d_in[19];  const float* b1 = (const float*)d_in[20];
  const float* W2 = (const float*)d_in[21];  const float* b2 = (const float*)d_in[22];
  const float* ln2_s = (const float*)d_in[23]; const float* ln2_b = (const float*)d_in[24];
  const float* pool_w = (const float*)d_in[25]; const float* pool_b = (const float*)d_in[26];

  // workspace bump allocator (total ~193 MB)
  char* ws = (char*)d_ws;
  size_t off = 0;
  auto alloc = [&](size_t bytes) -> void* {
    void* p = ws + off;
    off = (off + bytes + 255) & ~(size_t)255;
    return p;
  };
  float*  h      = (float*) alloc((size_t)kTok * kD * 4);
  __half* h16    = (__half*)alloc((size_t)kTok * kD * 2);
  __half* q16    = (__half*)alloc((size_t)kTok * kD * 2);
  __half* k16    = (__half*)alloc((size_t)kTok * kD * 2);
  __half* v16    = (__half*)alloc((size_t)kTok * kD * 2);
  __half* ctx16  = (__half*)alloc((size_t)kTok * kD * 2);
  float*  tmpf   = (float*) alloc((size_t)kTok * kD * 4);   // attn_out / ff2
  __half* ff1_16 = (__half*)alloc((size_t)kTok * kFF * 2);
  __half* Wq16   = (__half*)alloc((size_t)kNL * kD * kD * 2);
  __half* Wk16   = (__half*)alloc((size_t)kNL * kD * kD * 2);
  __half* Wv16   = (__half*)alloc((size_t)kNL * kD * kD * 2);
  __half* Wo16   = (__half*)alloc((size_t)kNL * kD * kD * 2);
  __half* W1_16  = (__half*)alloc((size_t)kNL * kD * kFF * 2);
  __half* W2_16  = (__half*)alloc((size_t)kNL * kFF * kD * 2);
  __half* Pw16   = (__half*)alloc((size_t)kD * kD * 2);
  int*    kb_idx = (int*)   alloc((size_t)kNB * kNB * 4);
  int*    kb_cnt = (int*)   alloc((size_t)kNB * 4);

  // weight conversion (re-done every call: deterministic, graph-capture safe)
  int nDD = kNL * kD * kD, nDF = kNL * kD * kFF;
  f32_to_f16_kernel<<<2048, 256, 0, stream>>>(Wq, Wq16, nDD);
  f32_to_f16_kernel<<<2048, 256, 0, stream>>>(Wk, Wk16, nDD);
  f32_to_f16_kernel<<<2048, 256, 0, stream>>>(Wv, Wv16, nDD);
  f32_to_f16_kernel<<<2048, 256, 0, stream>>>(Wo, Wo16, nDD);
  f32_to_f16_kernel<<<4096, 256, 0, stream>>>(W1, W1_16, nDF);
  f32_to_f16_kernel<<<4096, 256, 0, stream>>>(W2, W2_16, nDF);
  f32_to_f16_kernel<<<1024, 256, 0, stream>>>(pool_w, Pw16, kD * kD);
  build_kb_kernel<<<1, 64, 0, stream>>>(rand_blks, kb_idx, kb_cnt);

  embed_ln_kernel<<<kTok, 256, 0, stream>>>(input_ids, token_type, word_emb, type_emb,
                                            pos_emb, emb_ln_s, emb_ln_b, h, h16);

  dim3 gDD(kD / 128, kTok / 128);
  dim3 gFF1(kFF / 128, kTok / 128);

  for (int l = 0; l < kNL; ++l) {
    const __half* wq = Wq16 + (size_t)l * kD * kD;
    const __half* wk = Wk16 + (size_t)l * kD * kD;
    const __half* wv = Wv16 + (size_t)l * kD * kD;
    const __half* wo = Wo16 + (size_t)l * kD * kD;
    const __half* w1 = W1_16 + (size_t)l * kD * kFF;
    const __half* w2 = W2_16 + (size_t)l * kFF * kD;

    gemm_kernel<0, true ><<<gDD, 256, 0, stream>>>(h16, wq, bq + l * kD, q16, kTok, kD, kD, kD);
    gemm_kernel<0, true ><<<gDD, 256, 0, stream>>>(h16, wk, bk + l * kD, k16, kTok, kD, kD, kD);
    gemm_kernel<0, true ><<<gDD, 256, 0, stream>>>(h16, wv, bv + l * kD, v16, kTok, kD, kD, kD);

    attn_kernel<<<kB * kNH * kNB, 128, 0, stream>>>(q16, k16, v16, attn_mask,
                                                    kb_idx, kb_cnt, ctx16);

    gemm_kernel<0, false><<<gDD, 256, 0, stream>>>(ctx16, wo, bo + l * kD, tmpf, kTok, kD, kD, kD);
    add_ln_kernel<<<kTok, 256, 0, stream>>>(h, tmpf, ln1_s + l * kD, ln1_b + l * kD, h16);

    gemm_kernel<1, true ><<<gFF1, 256, 0, stream>>>(h16, w1, b1 + l * kFF, ff1_16, kTok, kFF, kD, kD);
    gemm_kernel<0, false><<<gDD, 256, 0, stream>>>(ff1_16, w2, b2 + l * kD, tmpf, kTok, kD, kFF, kFF);
    add_ln_kernel<<<kTok, 256, 0, stream>>>(h, tmpf, ln2_s + l * kD, ln2_b + l * kD, h16);
  }

  // pooler: rows are token 0 of each batch via lda = S*D; tanh epilogue -> d_out
  dim3 gPool(kD / 128, 1);
  gemm_kernel<2, false><<<gPool, 256, 0, stream>>>(h16, Pw16, pool_b, (float*)d_out,
                                                   kB, kD, kD, kS * kD);
}
